// MaskRCNN_13417477833168
// MI455X (gfx1250) — matrix operand order repaired and checked by reading, not compiled, not run on hardware
//
#include <hip/hip_runtime.h>
#include <hip/hip_bf16.h>
#include <math.h>

// ---------------------------------------------------------------------------
// Mask R-CNN forward for gfx1250 (MI455X).
// GEMMs via v_wmma_f32_16x16x32_f16 with fragment-major packed operands.
// Mask-head weights staged into LDS via the Tensor Data Mover (TDM).
// ---------------------------------------------------------------------------

typedef __attribute__((ext_vector_type(16))) _Float16 v16h;
typedef __attribute__((ext_vector_type(8)))  float    v8f;
typedef __attribute__((ext_vector_type(4)))  unsigned int u32x4;
typedef __attribute__((ext_vector_type(8)))  int      i32x8;
typedef __attribute__((ext_vector_type(4)))  int      i32x4;

#if defined(__has_builtin)
#if __has_builtin(__builtin_amdgcn_tensor_load_to_lds)
#define HAVE_TDM 1
#endif
#endif

#define NANCH 2304          // 16*16*9
#define NPROP 2000

// d_out flat offsets (float elements)
#define OFF_RPN_LOGITS   0
#define OFF_RPN_DELTAS   4608
#define OFF_PROPOSALS    13824
#define OFF_ANCHORS      21824
#define OFF_RCNN_LOGITS  31040
#define OFF_RCNN_DELTAS  35040
#define OFF_RCNN_MASKS   43040
#define OFF_FINAL_DETS   435040
#define OFF_FINAL_MASKS  443040
#define OFF_FINAL_SCORES 835040
#define OFF_KEEP2        837040

// ---------------------------------------------------------------------------
// Fragment-major layout helpers.
// A tile (mt,kc): 512 halves; lane l holds halves [l*16, l*16+16).
//   m = mt*16 + (l&15); k = kc*32 + ((l>>4)<<3) + (e&7) + ((e>>3)<<4)
// B tile (kc,nt): n = nt*16 + (l&15); k = kc*32 + ((l>>4)<<3) + (e&7) + ((e>>3)<<4)
// ---------------------------------------------------------------------------
__device__ __forceinline__ size_t pack_a_idx(int row, int col, int KCp) {
  int mt = row >> 4, ml = row & 15;
  int kc = col >> 5, r = col & 31;
  int lh = (r >> 3) & 1;
  int e  = (r & 7) | ((r >> 4) << 3);
  return (((size_t)mt * KCp + kc) << 9) + (size_t)(((lh << 4) | ml) << 4) + e;
}

// A fragment built on the fly from an LDS crop [64][14][14] (f16): im2col for a
// 3x3 SAME conv, k = c*9 + dy*3 + dx, pos = py*14+px (rows padded past 196).
__device__ __forceinline__ v16h frag_a_im2col(const _Float16* __restrict__ crop,
                                              int m0, int k0) {
  int lane = threadIdx.x & 31;
  int pos = m0 + (lane & 15);
  int kb  = k0 + ((lane >> 4) << 3);
  int py = pos / 14, px = pos % 14;
  bool okp = pos < 196;
  v16h a;
#pragma unroll
  for (int half = 0; half < 2; ++half) {
#pragma unroll
    for (int j = 0; j < 8; ++j) {
      int k = kb + half * 16 + j;
      int c = k / 9, r = k % 9;
      int dy = r / 3 - 1, dx = r % 3 - 1;
      int yy = py + dy, xx = px + dx;
      _Float16 v = (_Float16)0.0f;
      if (okp && yy >= 0 && yy < 14 && xx >= 0 && xx < 14)
        v = crop[c * 196 + yy * 14 + xx];
      a[half * 8 + j] = v;
    }
  }
  return a;
}

#if HAVE_TDM
// 2D TDM copy: `rows` rows of `rowDwords` dwords, global row stride
// `strideDwords` dwords, into contiguous LDS at byte offset lds_off.
__device__ __forceinline__ void tdm_load_2d(unsigned lds_off, const void* gptr,
                                            unsigned rowDwords, unsigned rows,
                                            unsigned strideDwords) {
  unsigned long long ga = (unsigned long long)gptr;
  u32x4 g0;
  g0[0] = 1u;                                        // count=1, user mode
  g0[1] = lds_off;                                   // lds_addr
  g0[2] = (unsigned)(ga & 0xFFFFFFFFull);            // global_addr lo
  g0[3] = (unsigned)((ga >> 32) & 0x1FFFFFFull) | (2u << 30);  // hi | type=2
  i32x8 g1;
  g1[0] = (int)(2u << 16);                           // data_size = 4B
  g1[1] = (int)((rowDwords & 0xFFFFu) << 16);        // tensor_dim0 lo
  g1[2] = (int)((rowDwords >> 16) | ((rows & 0xFFFFu) << 16));   // dim0 hi|dim1 lo
  g1[3] = (int)((rows >> 16) | ((rowDwords & 0xFFFFu) << 16));   // dim1 hi|tile0
  g1[4] = (int)(rows & 0xFFFFu);                     // tile_dim1 (tile_dim2=0)
  g1[5] = (int)strideDwords;                         // tensor_dim0_stride lo
  g1[6] = 0;
  g1[7] = 0;
  i32x4 z4 = {0, 0, 0, 0};
  i32x8 z8 = {0, 0, 0, 0, 0, 0, 0, 0};
  __builtin_amdgcn_tensor_load_to_lds(g0, g1, z4, z4, z8, 0);
}
#endif

// ---------------------------------------------------------------------------
// Generic WMMA GEMM over packed operands. One wave per 16x16 tile.
// Outputs: optional plain f32 (ldc), optional plain f16 (ldc), optional
// packed-A f16 for the next GEMM (packKC = nextK/32).
// ---------------------------------------------------------------------------
__global__ void gemm_wmma_pk(const _Float16* __restrict__ Ap,
                             const _Float16* __restrict__ Bp,
                             const float* __restrict__ bias,
                             float* __restrict__ Cf,
                             _Float16* __restrict__ Chp,
                             _Float16* __restrict__ Capk,
                             int M, int N, int K, int NT, int ldc,
                             int packKC, int relu) {
  int mt = blockIdx.x, nt = blockIdx.y;
  int lane = threadIdx.x & 31;
  int KC = K >> 5;
  const _Float16* arow = Ap + (((size_t)mt * KC) << 9) + (lane << 4);
  const _Float16* bcol = Bp + (((size_t)nt) << 9) + (lane << 4);
  v8f acc = {0.f, 0.f, 0.f, 0.f, 0.f, 0.f, 0.f, 0.f};
  for (int kc = 0; kc < KC; ++kc) {
    if (kc + 1 < KC)
      __builtin_prefetch(arow + ((size_t)(kc + 1) << 9), 0, 0);
    v16h a = *(const v16h*)(arow + ((size_t)kc << 9));
    v16h b = *(const v16h*)(bcol + (((size_t)kc * NT) << 9));
    acc = __builtin_amdgcn_wmma_f32_16x16x32_f16(false, a, false, b,
                                                 (short)0, acc, false, false);
  }
  int col = nt * 16 + (lane & 15);
  if (col < N) {
    float bv = bias ? bias[col] : 0.0f;
#pragma unroll
    for (int i = 0; i < 8; ++i) {
      int row = mt * 16 + i + ((lane >> 4) << 3);
      float v = acc[i] + bv;
      if (relu) v = fmaxf(v, 0.0f);
      if (Cf)   Cf[(size_t)row * ldc + col] = v;
      if (Chp)  Chp[(size_t)row * ldc + col] = (_Float16)v;
      if (Capk) Capk[pack_a_idx(row, col, packKC)] = (_Float16)v;
    }
  }
}

// ---------------------------------------------------------------------------
// Weight conversion f32 -> packed-B f16 tiles (pads n >= N with zeros).
// trans=1: source is [N,K] (conv OIHW flattened); trans=0: source is [K,N].
// ---------------------------------------------------------------------------
__global__ void convert_b_pk(const float* __restrict__ in,
                             _Float16* __restrict__ out,
                             int K, int N, int KC, int NT, int trans) {
  int i = blockIdx.x * blockDim.x + threadIdx.x;
  int total = KC * NT * 512;
  if (i >= total) return;
  int tile = i >> 9, w = i & 511, lane = w >> 4, e = w & 15;
  int kc = tile / NT, nt = tile % NT;
  int k = kc * 32 + ((lane >> 4) << 3) + (e & 7) + ((e >> 3) << 4);
  int n = nt * 16 + (lane & 15);
  float v = 0.0f;
  if (n < N) v = trans ? in[(size_t)n * K + k] : in[(size_t)k * N + n];
  out[i] = (_Float16)v;
}

// ---------------------------------------------------------------------------
// Packed im2col for backbone (16x16 stride-16 VALID conv on 3x256x256)
// ---------------------------------------------------------------------------
__global__ void im2col_bb_pk(const float* __restrict__ x,
                             _Float16* __restrict__ A) {
  int i = blockIdx.x * blockDim.x + threadIdx.x;
  if (i >= 16 * 24 * 512) return;
  int tile = i >> 9, w = i & 511, lane = w >> 4, e = w & 15;
  int mt = tile / 24, kc = tile % 24;
  int m = mt * 16 + (lane & 15);
  int k = kc * 32 + ((lane >> 4) << 3) + (e & 7) + ((e >> 3) << 4);
  int y = m / 16, xo = m % 16;
  int c = k / 256, r = k % 256, fy = r / 16, fx = r % 16;
  A[i] = (_Float16)x[(size_t)c * 65536 + (y * 16 + fy) * 256 + (xo * 16 + fx)];
}

// Packed im2col for RPN 3x3 SAME conv over featf [256,64] (pos-major, f32)
__global__ void im2col_rpn_pk(const float* __restrict__ featf,
                              _Float16* __restrict__ A) {
  int i = blockIdx.x * blockDim.x + threadIdx.x;
  if (i >= 16 * 18 * 512) return;
  int tile = i >> 9, w = i & 511, lane = w >> 4, e = w & 15;
  int mt = tile / 18, kc = tile % 18;
  int m = mt * 16 + (lane & 15);
  int k = kc * 32 + ((lane >> 4) << 3) + (e & 7) + ((e >> 3) << 4);
  int c = k / 9, rr = k % 9;
  int dy = rr / 3 - 1, dx = rr % 3 - 1;
  int y = m / 16 + dy, x = m % 16 + dx;
  _Float16 v = (_Float16)0.0f;
  if (y >= 0 && y < 16 && x >= 0 && x < 16)
    v = (_Float16)featf[(y * 16 + x) * 64 + c];
  A[i] = v;
}

// ---------------------------------------------------------------------------
// Anchors + softmax + decode + clip for the 2304 RPN boxes
// ---------------------------------------------------------------------------
__global__ void anchors_decode_kernel(const float* __restrict__ logits,
                                      const float* __restrict__ deltas,
                                      float* __restrict__ anchors_out,
                                      float* __restrict__ boxes,
                                      float* __restrict__ scores) {
  int i = blockIdx.x * blockDim.x + threadIdx.x;
  if (i >= NANCH) return;
  int pos = i / 9, a = i % 9;
  int y = pos / 16, x = pos % 16;
  int si = a / 3, ri = a % 3;
  float sv = (si == 0) ? 32.0f : (si == 1) ? 64.0f : 128.0f;
  float rv = (ri == 0) ? 0.5f : (ri == 1) ? 1.0f : 2.0f;
  float sq = sqrtf(rv);
  float wA = sv * sq, hA = sv / sq;
  float cx = (x + 0.5f) * 16.0f, cy = (y + 0.5f) * 16.0f;
  float a0 = cx - 0.5f * wA, a1 = cy - 0.5f * hA;
  float a2 = cx + 0.5f * wA, a3 = cy + 0.5f * hA;
  anchors_out[i * 4 + 0] = a0; anchors_out[i * 4 + 1] = a1;
  anchors_out[i * 4 + 2] = a2; anchors_out[i * 4 + 3] = a3;
  float l0 = logits[i * 2], l1 = logits[i * 2 + 1];
  float m = fmaxf(l0, l1);
  float e0 = expf(l0 - m), e1 = expf(l1 - m);
  scores[i] = e1 / (e0 + e1);
  float d0 = deltas[i * 4 + 0], d1 = deltas[i * 4 + 1];
  float d2 = deltas[i * 4 + 2], d3 = deltas[i * 4 + 3];
  float w = a2 - a0, h = a3 - a1;
  float ccx = a0 + 0.5f * w, ccy = a1 + 0.5f * h;
  float ncx = ccx + d0 * w, ncy = ccy + d1 * h;
  float nw = w * expf(d2), nh = h * expf(d3);
  const float lim = 255.0f;
  boxes[i * 4 + 0] = fminf(fmaxf(ncx - 0.5f * nw, 0.f), lim);
  boxes[i * 4 + 1] = fminf(fmaxf(ncy - 0.5f * nh, 0.f), lim);
  boxes[i * 4 + 2] = fminf(fmaxf(ncx + 0.5f * nw, 0.f), lim);
  boxes[i * 4 + 3] = fminf(fmaxf(ncy + 0.5f * nh, 0.f), lim);
}

// ---------------------------------------------------------------------------
// Greedy NMS: single workgroup bitonic sort + barrier-stepped suppression.
// ---------------------------------------------------------------------------
__global__ void nms_kernel(const float* __restrict__ boxes,
                           const float* __restrict__ scores,
                           int n, int sortN, float thr,
                           int* __restrict__ ordOut, int* __restrict__ keepOut) {
  __shared__ float skey[4096];
  __shared__ int   sidx[4096];
  __shared__ float sarea[4096];
  __shared__ unsigned char skeep[4096];
  int tid = threadIdx.x, bdim = blockDim.x;
  for (int i = tid; i < sortN; i += bdim) {
    skey[i] = (i < n) ? -scores[i] : 3.0e38f;
    sidx[i] = i;
  }
  __syncthreads();
  for (int kk = 2; kk <= sortN; kk <<= 1) {
    for (int jj = kk >> 1; jj > 0; jj >>= 1) {
      for (int i = tid; i < sortN; i += bdim) {
        int ixj = i ^ jj;
        if (ixj > i) {
          bool up = ((i & kk) == 0);
          float ki = skey[i], kj = skey[ixj];
          if ((ki > kj) == up) {
            skey[i] = kj; skey[ixj] = ki;
            int t2 = sidx[i]; sidx[i] = sidx[ixj]; sidx[ixj] = t2;
          }
        }
      }
      __syncthreads();
    }
  }
  for (int i = tid; i < n; i += bdim) {
    int oi = sidx[i];
    float x1 = boxes[oi * 4], y1 = boxes[oi * 4 + 1];
    float x2 = boxes[oi * 4 + 2], y2 = boxes[oi * 4 + 3];
    sarea[i] = fmaxf(x2 - x1, 0.f) * fmaxf(y2 - y1, 0.f);
    skeep[i] = 1;
  }
  __syncthreads();
  for (int i = 0; i < n; ++i) {
    if (skeep[i]) {
      int oi = sidx[i];
      float ax1 = boxes[oi * 4], ay1 = boxes[oi * 4 + 1];
      float ax2 = boxes[oi * 4 + 2], ay2 = boxes[oi * 4 + 3];
      float aarea = sarea[i];
      for (int j = i + 1 + tid; j < n; j += bdim) {
        if (skeep[j]) {
          int oj = sidx[j];
          float bx1 = boxes[oj * 4], by1 = boxes[oj * 4 + 1];
          float bx2 = boxes[oj * 4 + 2], by2 = boxes[oj * 4 + 3];
          float xx1 = fmaxf(ax1, bx1), yy1 = fmaxf(ay1, by1);
          float xx2 = fminf(ax2, bx2), yy2 = fminf(ay2, by2);
          float inter = fmaxf(xx2 - xx1, 0.f) * fmaxf(yy2 - yy1, 0.f);
          float iou = inter / (aarea + sarea[j] - inter + 1e-8f);
          if (iou > thr) skeep[j] = 0;
        }
      }
    }
    __syncthreads();
  }
  for (int i = tid; i < n; i += bdim) {
    ordOut[i] = sidx[i];
    keepOut[i] = (int)skeep[i];
  }
}

// ---------------------------------------------------------------------------
// Proposal selection: kept (score order) first, then suppressed; take 2000.
// ---------------------------------------------------------------------------
__global__ void proposals_select_kernel(const int* __restrict__ ord,
                                        const int* __restrict__ keep,
                                        const float* __restrict__ boxes,
                                        float* __restrict__ prop_out,
                                        float* __restrict__ nb,
                                        int* __restrict__ valid) {
  __shared__ int slot[NANCH];
  int tid = threadIdx.x;
  if (tid == 0) {
    int kc = 0;
    for (int i = 0; i < NANCH; ++i) kc += keep[i] ? 1 : 0;
    int a = 0, b = kc;
    for (int i = 0; i < NANCH; ++i) slot[i] = keep[i] ? a++ : b++;
  }
  __syncthreads();
  for (int i = tid; i < NANCH; i += blockDim.x) {
    int r = slot[i];
    if (r < NPROP) {
      int v = keep[i];
      int oi = ord[i];
      float b0 = v ? boxes[oi * 4 + 0] : 0.f;
      float b1 = v ? boxes[oi * 4 + 1] : 0.f;
      float b2 = v ? boxes[oi * 4 + 2] : 0.f;
      float b3 = v ? boxes[oi * 4 + 3] : 0.f;
      prop_out[r * 4 + 0] = b0; prop_out[r * 4 + 1] = b1;
      prop_out[r * 4 + 2] = b2; prop_out[r * 4 + 3] = b3;
      valid[r] = v;
      const float inv = 1.0f / 255.0f;
      nb[r * 4 + 0] = b1 * inv; nb[r * 4 + 1] = b0 * inv;
      nb[r * 4 + 2] = b3 * inv; nb[r * 4 + 3] = b2 * inv;
    }
  }
}

// ---------------------------------------------------------------------------
// Per-proposal: bilinear crop -> LDS, 2x2 maxpool -> packed flat f16 (for FC1),
// mask 3x3 conv via WMMA (B staged to LDS by TDM), m2 1x1 conv accumulated
// with LDS float atomics -> rcnn_masks.
// ---------------------------------------------------------------------------
__global__ void crop_mask_kernel(const float* __restrict__ featpos,
                                 const float* __restrict__ nb,
                                 const _Float16* __restrict__ wm1P,
                                 const float* __restrict__ b_m1,
                                 const float* __restrict__ w_m2,
                                 const float* __restrict__ b_m2,
                                 _Float16* __restrict__ flatP,
                                 float* __restrict__ masks_out) {
  __shared__ __align__(32) _Float16 crop[64 * 196];
  __shared__ __align__(32) _Float16 Bstage[18 * 512];
  __shared__ float maskacc[208];
  __shared__ float wyv[14], wxv[14];
  __shared__ int y0a[14], y1a[14], x0a[14], x1a[14];
  int p = blockIdx.x;
  int tid = threadIdx.x;
  if (tid < 14) {
    float t = (float)tid / 13.0f;
    float y1 = nb[p * 4 + 0], x1 = nb[p * 4 + 1];
    float y2 = nb[p * 4 + 2], x2 = nb[p * 4 + 3];
    float fy = (y1 + (y2 - y1) * t) * 15.0f;
    float fx = (x1 + (x2 - x1) * t) * 15.0f;
    float y0f = fminf(fmaxf(floorf(fy), 0.f), 15.f);
    float x0f = fminf(fmaxf(floorf(fx), 0.f), 15.f);
    int y0 = (int)y0f, x0 = (int)x0f;
    y0a[tid] = y0; x0a[tid] = x0;
    y1a[tid] = min(y0 + 1, 15); x1a[tid] = min(x0 + 1, 15);
    wyv[tid] = fy - y0f; wxv[tid] = fx - x0f;
  }
  for (int i = tid; i < 208; i += blockDim.x) maskacc[i] = b_m2[0];
  __syncthreads();
  for (int i = tid; i < 64 * 196; i += blockDim.x) {
    int c = i / 196, r = i % 196, py = r / 14, px = r % 14;
    float wx = wxv[px], wy = wyv[py];
    int yA = y0a[py], yB = y1a[py], xA = x0a[px], xB = x1a[px];
    float v00 = featpos[(yA * 16 + xA) * 64 + c];
    float v01 = featpos[(yA * 16 + xB) * 64 + c];
    float v10 = featpos[(yB * 16 + xA) * 64 + c];
    float v11 = featpos[(yB * 16 + xB) * 64 + c];
    float top = v00 * (1.f - wx) + v01 * wx;
    float bot = v10 * (1.f - wx) + v11 * wx;
    crop[i] = (_Float16)(top * (1.f - wy) + bot * wy);
  }
  __syncthreads();
  // 2x2 maxpool -> packed flat features for FC1 (row = p, K = 3136, KC = 98)
  for (int i = tid; i < 64 * 49; i += blockDim.x) {
    int c = i / 49, r = i % 49, oy = r / 7, ox = r % 7;
    const _Float16* cb = crop + c * 196;
    float m0 = fmaxf((float)cb[(2 * oy) * 14 + 2 * ox], (float)cb[(2 * oy) * 14 + 2 * ox + 1]);
    float m1 = fmaxf((float)cb[(2 * oy + 1) * 14 + 2 * ox], (float)cb[(2 * oy + 1) * 14 + 2 * ox + 1]);
    flatP[pack_a_idx(p, i, 98)] = (_Float16)fmaxf(m0, m1);
  }
  // mask conv: 196(pad 208) x 576 x 64; loop N tiles, stage B panel in LDS
  int wave = tid >> 5;
  int lane = tid & 31;
  for (int nt = 0; nt < 4; ++nt) {
    __syncthreads();   // all waves done with previous panel
#if HAVE_TDM
    if (tid < 32) {
      tdm_load_2d((unsigned)(size_t)&Bstage[0], wm1P + nt * 512,
                  256u, 18u, 1024u);
      __builtin_amdgcn_s_wait_tensorcnt((short)0);
    }
#else
    for (int i = tid; i < 18 * 512; i += blockDim.x) {
      int kc = i >> 9, j = i & 511;
      Bstage[i] = wm1P[((size_t)(kc * 4 + nt) << 9) + j];
    }
#endif
    __syncthreads();
    int col = nt * 16 + (lane & 15);
    float w2 = w_m2[col];
    float bm = b_m1[col];
    for (int t = wave; t < 13; t += 8) {
      v8f acc = {0.f, 0.f, 0.f, 0.f, 0.f, 0.f, 0.f, 0.f};
      for (int kc = 0; kc < 18; ++kc) {
        v16h a = frag_a_im2col(crop, t * 16, kc * 32);
        v16h b = *(const v16h*)(Bstage + (kc << 9) + (lane << 4));
        acc = __builtin_amdgcn_wmma_f32_16x16x32_f16(false, a, false, b,
                                                     (short)0, acc, false, false);
      }
#pragma unroll
      for (int i2 = 0; i2 < 8; ++i2) {
        int row = t * 16 + i2 + ((lane >> 4) << 3);
        if (row < 196) {
          float v = fmaxf(acc[i2] + bm, 0.0f);
          atomicAdd(&maskacc[row], v * w2);
        }
      }
    }
  }
  __syncthreads();
  for (int i = tid; i < 196; i += blockDim.x)
    masks_out[(size_t)p * 196 + i] = maskacc[i];
}

// ---------------------------------------------------------------------------
// RCNN heads: logits/deltas from h2 (plain f16), softmax, decode, clip
// ---------------------------------------------------------------------------
__global__ void heads_kernel(const _Float16* __restrict__ h2,
                             const float* __restrict__ w_rcls,
                             const float* __restrict__ b_rcls,
                             const float* __restrict__ w_rbox,
                             const float* __restrict__ b_rbox,
                             const float* __restrict__ proposals,
                             const int* __restrict__ valid,
                             float* __restrict__ logits_out,
                             float* __restrict__ deltas_out,
                             float* __restrict__ det_scores,
                             float* __restrict__ dets,
                             float* __restrict__ s2) {
  int t = blockIdx.x * blockDim.x + threadIdx.x;
  if (t >= NPROP) return;
  float acc0 = 0.f, acc1 = 0.f, acc2 = 0.f, acc3 = 0.f, acc4 = 0.f, acc5 = 0.f;
  const _Float16* hp = h2 + (size_t)t * 512;
  for (int k = 0; k < 512; ++k) {
    float hv = (float)hp[k];
    acc0 += hv * w_rcls[k * 2 + 0];
    acc1 += hv * w_rcls[k * 2 + 1];
    acc2 += hv * w_rbox[k * 4 + 0];
    acc3 += hv * w_rbox[k * 4 + 1];
    acc4 += hv * w_rbox[k * 4 + 2];
    acc5 += hv * w_rbox[k * 4 + 3];
  }
  float l0 = acc0 + b_rcls[0], l1 = acc1 + b_rcls[1];
  logits_out[t * 2 + 0] = l0; logits_out[t * 2 + 1] = l1;
  float d0 = acc2 + b_rbox[0], d1 = acc3 + b_rbox[1];
  float d2 = acc4 + b_rbox[2], d3 = acc5 + b_rbox[3];
  deltas_out[t * 4 + 0] = d0; deltas_out[t * 4 + 1] = d1;
  deltas_out[t * 4 + 2] = d2; deltas_out[t * 4 + 3] = d3;
  float m = fmaxf(l0, l1);
  float e0 = expf(l0 - m), e1 = expf(l1 - m);
  float sc = e1 / (e0 + e1);
  det_scores[t] = sc;
  float a0 = proposals[t * 4 + 0], a1 = proposals[t * 4 + 1];
  float a2 = proposals[t * 4 + 2], a3 = proposals[t * 4 + 3];
  float w = a2 - a0, h = a3 - a1;
  float ccx = a0 + 0.5f * w, ccy = a1 + 0.5f * h;
  float ncx = ccx + d0 * w, ncy = ccy + d1 * h;
  float nw = w * expf(d2), nh = h * expf(d3);
  const float lim = 255.0f;
  dets[t * 4 + 0] = fminf(fmaxf(ncx - 0.5f * nw, 0.f), lim);
  dets[t * 4 + 1] = fminf(fmaxf(ncy - 0.5f * nh, 0.f), lim);
  dets[t * 4 + 2] = fminf(fmaxf(ncx + 0.5f * nw, 0.f), lim);
  dets[t * 4 + 3] = fminf(fmaxf(ncy + 0.5f * nh, 0.f), lim);
  s2[t] = valid[t] ? sc : -1.0f;
}

// ---------------------------------------------------------------------------
// Finalize detections after second NMS
// ---------------------------------------------------------------------------
__global__ void finalize_kernel(const int* __restrict__ ord2,
                                const int* __restrict__ keep2raw,
                                const int* __restrict__ valid,
                                const float* __restrict__ dets,
                                const float* __restrict__ det_scores,
                                float* __restrict__ fdets,
                                float* __restrict__ fscores,
                                float* __restrict__ keep2f,
                                int* __restrict__ keep2fin) {
  int r = blockIdx.x * blockDim.x + threadIdx.x;
  if (r >= NPROP) return;
  int j = ord2[r];
  int k2 = (keep2raw[r] && valid[j]) ? 1 : 0;
  keep2fin[r] = k2;
  keep2f[r] = (float)k2;
  fscores[r] = k2 ? det_scores[j] : 0.0f;
#pragma unroll
  for (int c = 0; c < 4; ++c) fdets[r * 4 + c] = k2 ? dets[j * 4 + c] : 0.0f;
}

__global__ void final_masks_kernel(const int* __restrict__ ord2,
                                   const int* __restrict__ keep2fin,
                                   const float* __restrict__ masks_in,
                                   float* __restrict__ masks_out) {
  int r = blockIdx.x;
  int pos = threadIdx.x;
  if (pos >= 196) return;
  int j = ord2[r];
  float v = masks_in[(size_t)j * 196 + pos];
  float sg = 1.0f / (1.0f + expf(-v));
  masks_out[(size_t)r * 196 + pos] = sg * (float)keep2fin[r];
}

// ---------------------------------------------------------------------------
// Host driver
// ---------------------------------------------------------------------------
extern "C" void kernel_launch(void* const* d_in, const int* in_sizes, int n_in,
                              void* d_out, int out_size, void* d_ws, size_t ws_size,
                              hipStream_t stream) {
  (void)in_sizes; (void)n_in; (void)out_size; (void)ws_size;
  const float* x      = (const float*)d_in[0];
  const float* w_bb   = (const float*)d_in[1];
  const float* b_bb   = (const float*)d_in[2];
  const float* w_rpn  = (const float*)d_in[3];
  const float* b_rpn  = (const float*)d_in[4];
  const float* w_cls  = (const float*)d_in[5];
  const float* b_cls  = (const float*)d_in[6];
  const float* w_box  = (const float*)d_in[7];
  const float* b_box  = (const float*)d_in[8];
  const float* w_fc1  = (const float*)d_in[9];
  const float* b_fc1  = (const float*)d_in[10];
  const float* w_fc2  = (const float*)d_in[11];
  const float* b_fc2  = (const float*)d_in[12];
  const float* w_rcls = (const float*)d_in[13];
  const float* b_rcls = (const float*)d_in[14];
  const float* w_rbox = (const float*)d_in[15];
  const float* b_rbox = (const float*)d_in[16];
  const float* w_m1   = (const float*)d_in[17];
  const float* b_m1   = (const float*)d_in[18];
  const float* w_m2   = (const float*)d_in[19];
  const float* b_m2   = (const float*)d_in[20];
  float* out = (float*)d_out;

  char* ws = (char*)d_ws;
  size_t cur = 0;
  auto alloc = [&](size_t bytes) -> void* {
    void* p = ws + cur;
    cur += (bytes + 255) & ~(size_t)255;
    return p;
  };
  _Float16* AbbP     = (_Float16*)alloc((size_t)16 * 24 * 512 * 2);
  _Float16* wbbP     = (_Float16*)alloc((size_t)24 * 4 * 512 * 2);
  float*    featf    = (float*)   alloc((size_t)256 * 64 * 4);
  _Float16* ArpnP    = (_Float16*)alloc((size_t)16 * 18 * 512 * 2);
  _Float16* wrpnP    = (_Float16*)alloc((size_t)18 * 4 * 512 * 2);
  _Float16* hhP      = (_Float16*)alloc((size_t)16 * 2 * 512 * 2);
  _Float16* wclsP    = (_Float16*)alloc((size_t)2 * 2 * 512 * 2);
  _Float16* wboxP    = (_Float16*)alloc((size_t)2 * 3 * 512 * 2);
  _Float16* flatP    = (_Float16*)alloc((size_t)125 * 98 * 512 * 2);
  _Float16* wfc1P    = (_Float16*)alloc((size_t)98 * 32 * 512 * 2);
  _Float16* h1P      = (_Float16*)alloc((size_t)125 * 16 * 512 * 2);
  _Float16* wfc2P    = (_Float16*)alloc((size_t)16 * 32 * 512 * 2);
  _Float16* h2h      = (_Float16*)alloc((size_t)NPROP * 512 * 2);
  _Float16* wm1P     = (_Float16*)alloc((size_t)18 * 4 * 512 * 2);
  float*    boxes1   = (float*)alloc((size_t)NANCH * 4 * 4);
  float*    scores1  = (float*)alloc((size_t)NANCH * 4);
  float*    nb       = (float*)alloc((size_t)NPROP * 4 * 4);
  float*    dets     = (float*)alloc((size_t)NPROP * 4 * 4);
  float*    dscore   = (float*)alloc((size_t)NPROP * 4);
  float*    s2       = (float*)alloc((size_t)NPROP * 4);
  int*      ord1     = (int*)alloc((size_t)NANCH * 4);
  int*      keep1    = (int*)alloc((size_t)NANCH * 4);
  int*      valid    = (int*)alloc((size_t)NPROP * 4);
  int*      ord2     = (int*)alloc((size_t)NPROP * 4);
  int*      keep2raw = (int*)alloc((size_t)NPROP * 4);
  int*      keep2fin = (int*)alloc((size_t)NPROP * 4);

  auto blocks = [](int total) { return (total + 255) / 256; };
  // --- weight conversions (f32 -> packed-B f16 tiles) ---
  convert_b_pk<<<blocks(24 * 4 * 512), 256, 0, stream>>>(w_bb, wbbP, 768, 64, 24, 4, 1);
  convert_b_pk<<<blocks(18 * 4 * 512), 256, 0, stream>>>(w_rpn, wrpnP, 576, 64, 18, 4, 1);
  convert_b_pk<<<blocks(2 * 2 * 512), 256, 0, stream>>>(w_cls, wclsP, 64, 18, 2, 2, 1);
  convert_b_pk<<<blocks(2 * 3 * 512), 256, 0, stream>>>(w_box, wboxP, 64, 36, 2, 3, 1);
  convert_b_pk<<<blocks(98 * 32 * 512), 256, 0, stream>>>(w_fc1, wfc1P, 3136, 512, 98, 32, 0);
  convert_b_pk<<<blocks(16 * 32 * 512), 256, 0, stream>>>(w_fc2, wfc2P, 512, 512, 16, 32, 0);
  convert_b_pk<<<blocks(18 * 4 * 512), 256, 0, stream>>>(w_m1, wm1P, 576, 64, 18, 4, 1);

  // --- backbone: patchify GEMM 256x768x64 + relu -> featf (plain f32) ---
  im2col_bb_pk<<<blocks(16 * 24 * 512), 256, 0, stream>>>(x, AbbP);
  gemm_wmma_pk<<<dim3(16, 4), 32, 0, stream>>>(AbbP, wbbP, b_bb,
                                               featf, nullptr, nullptr,
                                               256, 64, 768, 4, 64, 0, 1);
  // --- RPN 3x3 conv: GEMM 256x576x64 + relu -> packed (KC'=2) ---
  im2col_rpn_pk<<<blocks(16 * 18 * 512), 256, 0, stream>>>(featf, ArpnP);
  gemm_wmma_pk<<<dim3(16, 4), 32, 0, stream>>>(ArpnP, wrpnP, b_rpn,
                                               nullptr, nullptr, hhP,
                                               256, 64, 576, 4, 0, 2, 1);
  // --- cls/box 1x1 convs straight into d_out ---
  gemm_wmma_pk<<<dim3(16, 2), 32, 0, stream>>>(hhP, wclsP, b_cls,
                                               out + OFF_RPN_LOGITS, nullptr, nullptr,
                                               256, 18, 64, 2, 18, 0, 0);
  gemm_wmma_pk<<<dim3(16, 3), 32, 0, stream>>>(hhP, wboxP, b_box,
                                               out + OFF_RPN_DELTAS, nullptr, nullptr,
                                               256, 36, 64, 3, 36, 0, 0);
  // --- anchors + softmax + decode + clip ---
  anchors_decode_kernel<<<9, 256, 0, stream>>>(out + OFF_RPN_LOGITS,
                                               out + OFF_RPN_DELTAS,
                                               out + OFF_ANCHORS, boxes1, scores1);
  // --- NMS 1 + proposal selection ---
  nms_kernel<<<1, 1024, 0, stream>>>(boxes1, scores1, NANCH, 4096, 0.5f, ord1, keep1);
  proposals_select_kernel<<<1, 1024, 0, stream>>>(ord1, keep1, boxes1,
                                                  out + OFF_PROPOSALS, nb, valid);
  // --- per-proposal crop / pool / mask head (TDM + WMMA from LDS) ---
  crop_mask_kernel<<<NPROP, 256, 0, stream>>>(featf, nb, wm1P, b_m1, w_m2, b_m2,
                                              flatP, out + OFF_RCNN_MASKS);
  // --- FC head GEMMs ---
  gemm_wmma_pk<<<dim3(125, 32), 32, 0, stream>>>(flatP, wfc1P, b_fc1,
                                                 nullptr, nullptr, h1P,
                                                 NPROP, 512, 3136, 32, 0, 16, 1);
  gemm_wmma_pk<<<dim3(125, 32), 32, 0, stream>>>(h1P, wfc2P, b_fc2,
                                                 nullptr, h2h, nullptr,
                                                 NPROP, 512, 512, 32, 512, 0, 1);
  // --- logits/deltas/scores/dets ---
  heads_kernel<<<8, 256, 0, stream>>>(h2h, w_rcls, b_rcls, w_rbox, b_rbox,
                                      out + OFF_PROPOSALS, valid,
                                      out + OFF_RCNN_LOGITS, out + OFF_RCNN_DELTAS,
                                      dscore, dets, s2);
  // --- NMS 2 + finalize ---
  nms_kernel<<<1, 1024, 0, stream>>>(dets, s2, NPROP, 2048, 0.3f, ord2, keep2raw);
  finalize_kernel<<<8, 256, 0, stream>>>(ord2, keep2raw, valid, dets, dscore,
                                         out + OFF_FINAL_DETS, out + OFF_FINAL_SCORES,
                                         out + OFF_KEEP2, keep2fin);
  final_masks_kernel<<<NPROP, 224, 0, stream>>>(ord2, keep2fin,
                                                out + OFF_RCNN_MASKS,
                                                out + OFF_FINAL_MASKS);
}
